// get_model_53017076302319
// MI455X (gfx1250) — matrix operand order repaired and checked by reading, not compiled, not run on hardware
//
#include <hip/hip_runtime.h>

typedef _Float16 half_t;
typedef __attribute__((ext_vector_type(16))) _Float16 v16h;
typedef __attribute__((ext_vector_type(8)))  _Float16 v8h;
typedef __attribute__((ext_vector_type(8)))  float    v8f;
typedef __attribute__((ext_vector_type(4)))  float    v4f;
typedef __attribute__((ext_vector_type(4)))  int      v4i;
typedef __attribute__((address_space(1))) v4i* gptr128_t;  // global v4i*
typedef __attribute__((address_space(3))) v4i* lptr128_t;  // LDS v4i*

#define BATCH   4
#define NPTS    4096
#define NCENT   512
#define NROWS   (BATCH * NCENT)   // 2048
#define SA3_CIN_PAD 352           // 323 padded to mult of 32
#define SA3_CIN_REAL 323

// ---------------------------------------------------------------------------
// Kernel 1: split (B,6,N) xyz into coords (B,N,3) and feats (B,N,3), f32.
// ---------------------------------------------------------------------------
__global__ void k_prep(const float* __restrict__ xyz,
                       float* __restrict__ coords, float* __restrict__ feats) {
  int i = blockIdx.x * 256 + threadIdx.x;
  if (i >= BATCH * NPTS) return;
  int b = i >> 12, n = i & (NPTS - 1);
  const float* src = xyz + (size_t)b * 6 * NPTS;
  float* cd = coords + (size_t)i * 3;
  float* ft = feats  + (size_t)i * 3;
  cd[0] = src[0 * NPTS + n]; cd[1] = src[1 * NPTS + n]; cd[2] = src[2 * NPTS + n];
  ft[0] = src[3 * NPTS + n]; ft[1] = src[4 * NPTS + n]; ft[2] = src[5 * NPTS + n];
}

// ---------------------------------------------------------------------------
// Kernel 2: pack a layer's weights (f32, [cout][cinReal]) into f16 B-fragment
// order: Wp[(nt*KT + kt)*32 + lane][e] = W[nt*16 + (lane&15)]
//                                          [kt*32 + (lane<16?0:8) + e + (e<8?0:8)]
// zero-padded for k >= cinReal. One thread per (tile, lane).
// ---------------------------------------------------------------------------
__global__ void k_packw(const float* __restrict__ W, half_t* __restrict__ Wp,
                        int cinPad, int cinReal, int cout) {
  int ktiles = cinPad >> 5;
  int total = (cout >> 4) * ktiles * 32;
  int i = blockIdx.x * 256 + threadIdx.x;
  if (i >= total) return;
  int lane = i & 31;
  int tile = i >> 5;
  int kt = tile % ktiles, nt = tile / ktiles;
  int n = nt * 16 + (lane & 15);
  int kbase = kt * 32 + ((lane < 16) ? 0 : 8);
  half_t* dst = Wp + (size_t)i * 16;
  for (int e = 0; e < 16; ++e) {
    int k = kbase + e + ((e < 8) ? 0 : 8);
    dst[e] = (k < cinReal) ? (half_t)W[(size_t)n * cinReal + k] : (half_t)0;
  }
}

// ---------------------------------------------------------------------------
// Kernel 3: farthest point sampling. One workgroup (1024 thr = 32 waves) per
// batch; 4 points per thread register-resident; LDS tree argmax per step.
// ---------------------------------------------------------------------------
__global__ __launch_bounds__(1024) void k_fps(const float* __restrict__ coords,
                                              int* __restrict__ fps_idx,
                                              float* __restrict__ new_xyz) {
  __shared__ float s_cent[3];
  __shared__ float s_val[1024];
  __shared__ int   s_idx[1024];
  int b = blockIdx.x;
  int t = threadIdx.x;
  const float* C = coords + (size_t)b * NPTS * 3;
  float px[4], py[4], pz[4], dist[4];
#pragma unroll
  for (int i = 0; i < 4; ++i) {
    int p = t * 4 + i;
    px[i] = C[p * 3 + 0]; py[i] = C[p * 3 + 1]; pz[i] = C[p * 3 + 2];
    dist[i] = 1e10f;
  }
  int far = 0;
  for (int it = 0; it < NCENT; ++it) {
    if (t == 0) {
      fps_idx[b * NCENT + it] = far;
      float cx = C[far * 3 + 0], cy = C[far * 3 + 1], cz = C[far * 3 + 2];
      s_cent[0] = cx; s_cent[1] = cy; s_cent[2] = cz;
      float* q = new_xyz + ((size_t)b * NCENT + it) * 3;
      q[0] = cx; q[1] = cy; q[2] = cz;
    }
    __syncthreads();
    float cx = s_cent[0], cy = s_cent[1], cz = s_cent[2];
    float best = -1.0f; int bi = 0;
#pragma unroll
    for (int i = 0; i < 4; ++i) {
      float dx = px[i] - cx, dy = py[i] - cy, dz = pz[i] - cz;
      float d = dx * dx + dy * dy + dz * dz;
      dist[i] = dist[i] < d ? dist[i] : d;
      if (dist[i] > best) { best = dist[i]; bi = t * 4 + i; }
    }
    s_val[t] = best; s_idx[t] = bi;
    __syncthreads();
    for (int s = 512; s > 0; s >>= 1) {
      if (t < s && s_val[t + s] > s_val[t]) { s_val[t] = s_val[t + s]; s_idx[t] = s_idx[t + s]; }
      __syncthreads();
    }
    far = s_idx[0];
  }
}

// ---------------------------------------------------------------------------
// Kernel 4: ball query. Coords for the batch staged into LDS with CDNA5 async
// global->LDS copies when available; one thread per center scans serially to
// preserve "first K in ascending index order" semantics.
// ---------------------------------------------------------------------------
__global__ __launch_bounds__(256) void k_ball(const float* __restrict__ coords,
                                              const float* __restrict__ new_xyz,
                                              int* __restrict__ gidx,
                                              float r2, int K) {
  __shared__ __attribute__((aligned(32))) float sc[NPTS * 3];
  int b  = blockIdx.x >> 1;
  int c0 = (blockIdx.x & 1) * 256;
  const float* C = coords + (size_t)b * NPTS * 3;
#if __has_builtin(__builtin_amdgcn_global_load_async_to_lds_b128)
  // 48KB contiguous stage-in: 16B per lane per op, tracked by ASYNCcnt.
  for (int i = threadIdx.x * 4; i < NPTS * 3; i += 256 * 4) {
    gptr128_t gp = (gptr128_t)(unsigned long long)(const void*)(C + i);
    lptr128_t lp = (lptr128_t)(unsigned)(unsigned long long)(void*)(sc + i);
    __builtin_amdgcn_global_load_async_to_lds_b128(gp, lp, 0, 0);
  }
#if __has_builtin(__builtin_amdgcn_s_wait_asynccnt)
  __builtin_amdgcn_s_wait_asynccnt(0);
#else
  asm volatile("s_wait_asynccnt 0x0" ::: "memory");
#endif
#else
  for (int i = threadIdx.x * 4; i < NPTS * 3; i += 256 * 4)
    *(v4f*)(sc + i) = *(const v4f*)(C + i);
#endif
  __syncthreads();
  int c = c0 + threadIdx.x;
  const float* q = new_xyz + ((size_t)b * NCENT + c) * 3;
  float qx = q[0], qy = q[1], qz = q[2];
  int* out = gidx + ((size_t)b * NCENT + c) * K;
  int cnt = 0;
  for (int n = 0; n < NPTS && cnt < K; ++n) {
    float dx = sc[n * 3 + 0] - qx, dy = sc[n * 3 + 1] - qy, dz = sc[n * 3 + 2] - qz;
    if (dx * dx + dy * dy + dz * dz <= r2) out[cnt++] = n;
  }
  int f = (cnt > 0) ? out[0] : 0;
  for (; cnt < K; ++cnt) out[cnt] = f;
}

// ---------------------------------------------------------------------------
// WMMA layer over LDS-resident activations, compile-time shapes.
// A fragment = two contiguous ds_load_b128 per k-tile (ISA 7.12.2 layout);
// B fragment = one 32B load from pre-packed f16 weights.
// ---------------------------------------------------------------------------
template <int ROWS, int CINPAD, int COUT>
__device__ __forceinline__ void mlp_layer_lds_t(
    const half_t* sIn, half_t* sOut, const half_t* __restrict__ Wp,
    const float* __restrict__ scale, const float* __restrict__ shift,
    int wave, int lane) {
  constexpr int MT = ROWS / 16, NT = COUT / 16, KT = CINPAD / 32;
  constexpr int NTOT = MT * NT;
  int kh = (lane < 16) ? 0 : 8;
  for (int tile = wave; tile < NTOT; tile += 4) {
    int mt = tile / NT, nt = tile % NT;
    int row = mt * 16 + (lane & 15);
    int n   = nt * 16 + (lane & 15);
    const half_t* xrow = sIn + (size_t)row * CINPAD;
    const half_t* wl   = Wp + ((size_t)nt * KT * 32 + lane) * 16;
    v8f acc = {};
#pragma unroll
    for (int kt = 0; kt < KT; ++kt) {
      v8h a0 = *(const v8h*)(xrow + kt * 32 + kh);
      v8h a1 = *(const v8h*)(xrow + kt * 32 + 16 + kh);
      v16h a = __builtin_shufflevector(a0, a1, 0, 1, 2, 3, 4, 5, 6, 7,
                                       8, 9, 10, 11, 12, 13, 14, 15);
      v16h bfrag = *(const v16h*)(wl + (size_t)kt * 32 * 16);
      acc = __builtin_amdgcn_wmma_f32_16x16x32_f16(
          false, a, false, bfrag, (short)0, acc, false, false);
    }
    float sc = scale[n], sh = shift[n];
#pragma unroll
    for (int v = 0; v < 8; ++v) {
      int r = mt * 16 + v + ((lane < 16) ? 0 : 8);
      float y = acc[v] * sc + sh;
      sOut[(size_t)r * COUT + n] = (half_t)(y > 0.f ? y : 0.f);
    }
  }
}

// ---------------------------------------------------------------------------
// Kernel 5: fused per-center branch MLP (gather -> 3 WMMA layers in LDS ->
// max over samples -> f16 into SA3 input buffer). One WG (4 waves) per center.
// ---------------------------------------------------------------------------
template <int NS, int W0, int W1, int W2>
__global__ __launch_bounds__(128) void k_branch_mlp_t(
    const float* __restrict__ coords, const float* __restrict__ feats,
    const float* __restrict__ new_xyz, const int* __restrict__ gidx,
    const half_t* __restrict__ Wp0, const float* __restrict__ s0, const float* __restrict__ h0,
    const half_t* __restrict__ Wp1, const float* __restrict__ s1, const float* __restrict__ h1,
    const half_t* __restrict__ Wp2, const float* __restrict__ s2, const float* __restrict__ h2,
    half_t* __restrict__ Xs, int chOff) {
  constexpr int WA = (W1 > 32) ? W1 : 32;   // sA widths: gather(32) / W1
  constexpr int WB = (W2 > W0) ? W2 : W0;   // sB widths: W0 / W2
  __shared__ __attribute__((aligned(32))) half_t sA[NS * WA];
  __shared__ __attribute__((aligned(32))) half_t sB[NS * WB];
  int cid = blockIdx.x;              // b*512 + c
  int b = cid >> 9;
  int tid = threadIdx.x, wave = tid >> 5, lane = tid & 31;
  const float* Q = new_xyz + (size_t)cid * 3;
  float qx = Q[0], qy = Q[1], qz = Q[2];
  const int* gi = gidx + (size_t)cid * NS;
  // gather: NS rows x 32 channels (feat0..2, rel xyz 3..5, zero pad 6..31)
  for (int r = tid; r < NS; r += 128) {
    int g = gi[r];
    const float* P = coords + ((size_t)b * NPTS + g) * 3;
    const float* F = feats  + ((size_t)b * NPTS + g) * 3;
    half_t* x = sA + (size_t)r * 32;
    x[0] = (half_t)F[0]; x[1] = (half_t)F[1]; x[2] = (half_t)F[2];
    x[3] = (half_t)(P[0] - qx); x[4] = (half_t)(P[1] - qy); x[5] = (half_t)(P[2] - qz);
#pragma unroll
    for (int k = 6; k < 32; ++k) x[k] = (half_t)0;
  }
  __syncthreads();
  mlp_layer_lds_t<NS, 32, W0>(sA, sB, Wp0, s0, h0, wave, lane);
  __syncthreads();
  mlp_layer_lds_t<NS, W0, W1>(sB, sA, Wp1, s1, h1, wave, lane);
  __syncthreads();
  mlp_layer_lds_t<NS, W1, W2>(sA, sB, Wp2, s2, h2, wave, lane);
  __syncthreads();
  // max-pool over NS samples, write into SA3 input (3 xyz channels first)
  half_t* xout = Xs + (size_t)cid * SA3_CIN_PAD + 3 + chOff;
  for (int ch = tid; ch < W2; ch += 128) {
    float m = 0.f;  // relu output >= 0
#pragma unroll 4
    for (int r = 0; r < NS; ++r) {
      float v = (float)sB[(size_t)r * W2 + ch];
      m = v > m ? v : m;
    }
    xout[ch] = (half_t)m;
  }
}

// ---------------------------------------------------------------------------
// Kernel 6: SA3 input fixup: xyz channels 0..2 and zero pad 323..351.
// ---------------------------------------------------------------------------
__global__ void k_sa3init(const float* __restrict__ new_xyz, half_t* __restrict__ Xs) {
  int r = blockIdx.x * 256 + threadIdx.x;
  if (r >= NROWS) return;
  half_t* x = Xs + (size_t)r * SA3_CIN_PAD;
  const float* q = new_xyz + (size_t)r * 3;
  x[0] = (half_t)q[0]; x[1] = (half_t)q[1]; x[2] = (half_t)q[2];
  for (int k = SA3_CIN_REAL; k < SA3_CIN_PAD; ++k) x[k] = (half_t)0;
}

// ---------------------------------------------------------------------------
// Kernel 7: generic global-activation WMMA MLP layer, compile-time shapes.
// One wave (block of 32) per 16x16 tile. Grid = (R/16, COUT/16).
// ---------------------------------------------------------------------------
template <int CINPAD, int COUT>
__global__ __launch_bounds__(32) void k_wmma_mlp_t(
    const half_t* __restrict__ X, const half_t* __restrict__ Wp,
    const float* __restrict__ scale, const float* __restrict__ shift,
    half_t* __restrict__ Y) {
  constexpr int KT = CINPAD / 32;
  int lane  = threadIdx.x;
  int mbase = blockIdx.x * 16;
  int nt    = blockIdx.y;
  int row = mbase + (lane & 15);
  int n   = nt * 16 + (lane & 15);
  int kh = (lane < 16) ? 0 : 8;
  const half_t* xrow  = X + (size_t)row * CINPAD;
  const half_t* wbase = Wp + ((size_t)nt * KT * 32 + lane) * 16;
  v8f acc = {};
#pragma unroll
  for (int kt = 0; kt < KT; ++kt) {
    v8h a0 = *(const v8h*)(xrow + kt * 32 + kh);
    v8h a1 = *(const v8h*)(xrow + kt * 32 + 16 + kh);
    v16h a = __builtin_shufflevector(a0, a1, 0, 1, 2, 3, 4, 5, 6, 7,
                                     8, 9, 10, 11, 12, 13, 14, 15);
    v16h bfrag = *(const v16h*)(wbase + (size_t)kt * 32 * 16);
    acc = __builtin_amdgcn_wmma_f32_16x16x32_f16(
        false, a, false, bfrag, (short)0, acc, false, false);
  }
  float sc = scale[n], sh = shift[n];
#pragma unroll
  for (int v = 0; v < 8; ++v) {
    int r = mbase + v + ((lane < 16) ? 0 : 8);
    float y = acc[v] * sc + sh;
    Y[(size_t)r * COUT + n] = (half_t)(y > 0.f ? y : 0.f);
  }
}

// ---------------------------------------------------------------------------
// Kernel 8: final max over the 512 points of each batch -> (4,512) f32 out.
// ---------------------------------------------------------------------------
__global__ void k_outmax(const half_t* __restrict__ Y, float* __restrict__ out) {
  int i = blockIdx.x * 256 + threadIdx.x;
  if (i >= BATCH * 512) return;
  int b = i >> 9, ch = i & 511;
  const half_t* y = Y + (size_t)b * NCENT * 512 + ch;
  float m = 0.f;
#pragma unroll 4
  for (int p = 0; p < NCENT; ++p) {
    float v = (float)y[(size_t)p * 512];
    m = v > m ? v : m;
  }
  out[i] = m;
}

// ---------------------------------------------------------------------------
extern "C" void kernel_launch(void* const* d_in, const int* in_sizes, int n_in,
                              void* d_out, int out_size, void* d_ws, size_t ws_size,
                              hipStream_t stream) {
  (void)in_sizes; (void)n_in; (void)out_size; (void)ws_size;
  const float* xyz = (const float*)d_in[0];
  // msg_params: branch i, layer j -> (W, scale, shift) at 1 + i*9 + j*3
  auto msgW = [&](int i, int j) { return (const float*)d_in[1 + i * 9 + j * 3 + 0]; };
  auto msgS = [&](int i, int j) { return (const float*)d_in[1 + i * 9 + j * 3 + 1]; };
  auto msgH = [&](int i, int j) { return (const float*)d_in[1 + i * 9 + j * 3 + 2]; };
  auto sa3W = [&](int j) { return (const float*)d_in[28 + j * 3 + 0]; };
  auto sa3S = [&](int j) { return (const float*)d_in[28 + j * 3 + 1]; };
  auto sa3H = [&](int j) { return (const float*)d_in[28 + j * 3 + 2]; };

  // workspace carve-up (256B aligned), total ~8 MB
  char* ws = (char*)d_ws;
  size_t off = 0;
  auto carve = [&](size_t bytes) -> char* {
    char* p = ws + off;
    off = (off + bytes + 255) & ~(size_t)255;
    return p;
  };
  float*  coords  = (float*)carve((size_t)BATCH * NPTS * 3 * 4);
  float*  feats   = (float*)carve((size_t)BATCH * NPTS * 3 * 4);
  int*    fpsIdx  = (int*)carve((size_t)BATCH * NCENT * 4);
  float*  newXyz  = (float*)carve((size_t)BATCH * NCENT * 3 * 4);
  int*    gidx0   = (int*)carve((size_t)NROWS * 16 * 4);
  int*    gidx1   = (int*)carve((size_t)NROWS * 32 * 4);
  int*    gidx2   = (int*)carve((size_t)NROWS * 128 * 4);
  half_t* Xs      = (half_t*)carve((size_t)NROWS * SA3_CIN_PAD * 2);
  half_t* Y1      = (half_t*)carve((size_t)NROWS * 128 * 2);
  half_t* Y2      = (half_t*)carve((size_t)NROWS * 256 * 2);
  half_t* Y3      = (half_t*)carve((size_t)NROWS * 512 * 2);

  // packed f16 weight buffers: [layer cinPad][cout]
  // msg: b0 (32,32)(32,32)(32,64)  b1/b2 (32,64)(64,64)(64,128)
  // sa3: (352,128)(128,256)(256,512)
  const int mCin[3][3]  = {{32, 32, 32}, {32, 64, 64}, {32, 64, 64}};
  const int mCinR[3][3] = {{6, 32, 32}, {6, 64, 64}, {6, 64, 64}};
  const int mCout[3][3] = {{32, 32, 64}, {64, 64, 128}, {64, 64, 128}};
  half_t* mWp[3][3];
  for (int i = 0; i < 3; ++i)
    for (int j = 0; j < 3; ++j)
      mWp[i][j] = (half_t*)carve((size_t)mCin[i][j] * mCout[i][j] * 2);
  const int sCin[3]  = {SA3_CIN_PAD, 128, 256};
  const int sCinR[3] = {SA3_CIN_REAL, 128, 256};
  const int sCout[3] = {128, 256, 512};
  half_t* sWp[3];
  for (int j = 0; j < 3; ++j)
    sWp[j] = (half_t*)carve((size_t)sCin[j] * sCout[j] * 2);

  // 1) layout prep
  k_prep<<<(BATCH * NPTS + 255) / 256, 256, 0, stream>>>(xyz, coords, feats);
  // 2) pack weights to f16 B-fragment layout (independent of everything else)
  for (int i = 0; i < 3; ++i)
    for (int j = 0; j < 3; ++j) {
      int tot = (mCout[i][j] >> 4) * (mCin[i][j] >> 5) * 32;
      k_packw<<<(tot + 255) / 256, 256, 0, stream>>>(
          msgW(i, j), mWp[i][j], mCin[i][j], mCinR[i][j], mCout[i][j]);
    }
  for (int j = 0; j < 3; ++j) {
    int tot = (sCout[j] >> 4) * (sCin[j] >> 5) * 32;
    k_packw<<<(tot + 255) / 256, 256, 0, stream>>>(
        sa3W(j), sWp[j], sCin[j], sCinR[j], sCout[j]);
  }
  // 3) FPS (one WG per batch)
  k_fps<<<BATCH, 1024, 0, stream>>>(coords, fpsIdx, newXyz);
  // 4) ball queries (async LDS stage-in)
  k_ball<<<BATCH * 2, 256, 0, stream>>>(coords, newXyz, gidx0, 0.1f * 0.1f, 16);
  k_ball<<<BATCH * 2, 256, 0, stream>>>(coords, newXyz, gidx1, 0.2f * 0.2f, 32);
  k_ball<<<BATCH * 2, 256, 0, stream>>>(coords, newXyz, gidx2, 0.4f * 0.4f, 128);
  // 5) SA3 input xyz + pad
  k_sa3init<<<(NROWS + 255) / 256, 256, 0, stream>>>(newXyz, Xs);
  // 6) fused per-center branch MLPs (WMMA in LDS) -> Xs channels [3, 323)
  k_branch_mlp_t<16, 32, 32, 64><<<NROWS, 128, 0, stream>>>(
      coords, feats, newXyz, gidx0,
      mWp[0][0], msgS(0, 0), msgH(0, 0), mWp[0][1], msgS(0, 1), msgH(0, 1),
      mWp[0][2], msgS(0, 2), msgH(0, 2), Xs, 0);
  k_branch_mlp_t<32, 64, 64, 128><<<NROWS, 128, 0, stream>>>(
      coords, feats, newXyz, gidx1,
      mWp[1][0], msgS(1, 0), msgH(1, 0), mWp[1][1], msgS(1, 1), msgH(1, 1),
      mWp[1][2], msgS(1, 2), msgH(1, 2), Xs, 64);
  k_branch_mlp_t<128, 64, 64, 128><<<NROWS, 128, 0, stream>>>(
      coords, feats, newXyz, gidx2,
      mWp[2][0], msgS(2, 0), msgH(2, 0), mWp[2][1], msgS(2, 1), msgH(2, 1),
      mWp[2][2], msgS(2, 2), msgH(2, 2), Xs, 192);
  // 7) SA3 MLP: 323(pad352) -> 128 -> 256 -> 512
  k_wmma_mlp_t<SA3_CIN_PAD, 128><<<dim3(NROWS / 16, 128 / 16), 32, 0, stream>>>(
      Xs, sWp[0], sa3S(0), sa3H(0), Y1);
  k_wmma_mlp_t<128, 256><<<dim3(NROWS / 16, 256 / 16), 32, 0, stream>>>(
      Y1, sWp[1], sa3S(1), sa3H(1), Y2);
  k_wmma_mlp_t<256, 512><<<dim3(NROWS / 16, 512 / 16), 32, 0, stream>>>(
      Y2, sWp[2], sa3S(2), sa3H(2), Y3);
  // 8) max over the 512 centers per batch -> (4,512) f32
  k_outmax<<<(BATCH * 512 + 255) / 256, 256, 0, stream>>>(Y3, (float*)d_out);
}
